// ResGCN_20289425506396
// MI455X (gfx1250) — matrix operand order repaired and checked by reading, not compiled, not run on hardware
//
#include <hip/hip_runtime.h>

typedef _Float16 half_t;
typedef __attribute__((ext_vector_type(16))) _Float16 v16h;
typedef __attribute__((ext_vector_type(8)))  _Float16 v8h;
typedef __attribute__((ext_vector_type(8)))  float    v8f;
typedef __attribute__((ext_vector_type(4)))  float    f4;
typedef __attribute__((ext_vector_type(4)))  unsigned int v4u;
typedef __attribute__((ext_vector_type(8)))  int      v8i;
typedef __attribute__((ext_vector_type(4)))  int      v4i;

#define HCH 128      // H channels
#define HIDC 256     // hidden channels
#define NGR 256      // num graphs
#define NCL 2        // num classes
#define EPS_MSG 1e-7f
#define EPS_SM  1e-16f
#define LN_EPS  1e-5f
#define MROWS 32     // rows per MLP block

// ---------------- utility kernels ----------------
__global__ void zero_i32(int* p, int n) {
    int i = blockIdx.x * blockDim.x + threadIdx.x;
    if (i < n) p[i] = 0;
}
__global__ void zero_f32(float* p, int n) {
    int i = blockIdx.x * blockDim.x + threadIdx.x;
    if (i < n) p[i] = 0.0f;
}

// h = x @ enc_W + enc_b   ([N,7] @ [7,128])
__global__ void encoder_kernel(const float* __restrict__ x, const float* __restrict__ W,
                               const float* __restrict__ b, float* __restrict__ h, int n) {
    int i = blockIdx.x * blockDim.x + threadIdx.x;
    if (i >= n * HCH) return;
    int node = i >> 7, c = i & (HCH - 1);
    float acc = b[c];
#pragma unroll
    for (int k = 0; k < 7; k++) acc += x[node * 7 + k] * W[k * HCH + c];
    h[i] = acc;
}

// ---------------- CSR-by-dst build ----------------
__global__ void count_kernel(const int* __restrict__ dst, int* __restrict__ cnt, int e) {
    int i = blockIdx.x * blockDim.x + threadIdx.x;
    if (i < e) atomicAdd(&cnt[dst[i]], 1);
}

__global__ __launch_bounds__(1024) void scan_kernel(const int* __restrict__ cnt,
                                                    int* __restrict__ rowptr, int n) {
    __shared__ int sm[1024];
    int tid = threadIdx.x;
    int chunk = (n + 1023) >> 10;
    int beg = tid * chunk;
    int end = beg + chunk; if (end > n) end = n;
    int s = 0;
    for (int i = beg; i < end; i++) s += cnt[i];
    sm[tid] = s;
    __syncthreads();
    for (int off = 1; off < 1024; off <<= 1) {
        int v = (tid >= off) ? sm[tid - off] : 0;
        __syncthreads();
        sm[tid] += v;
        __syncthreads();
    }
    int run = sm[tid] - s;   // exclusive prefix for this thread's chunk
    for (int i = beg; i < end; i++) { rowptr[i] = run; run += cnt[i]; }
    if (tid == 1023) rowptr[n] = sm[1023];
}

__global__ void fill_kernel(const int* __restrict__ src, const int* __restrict__ dst,
                            const int* __restrict__ rowptr, int* __restrict__ cur,
                            int* __restrict__ csrc, int e) {
    int i = blockIdx.x * blockDim.x + threadIdx.x;
    if (i >= e) return;
    int d = dst[i];
    int pos = rowptr[d] + atomicAdd(&cur[d], 1);
    csrc[pos] = src[i];
}

// ---------------- pre-message LayerNorm + ReLU (wave per node) ----------------
__global__ __launch_bounds__(256) void znorm_kernel(const float* __restrict__ h, float* __restrict__ z,
                                                    const float* __restrict__ g, const float* __restrict__ b,
                                                    int n) {
    int gid = blockIdx.x * blockDim.x + threadIdx.x;
    int node = gid >> 5;
    if (node >= n) return;
    int lane = gid & 31;
    const f4* h4 = (const f4*)h;
    f4 hv = h4[node * 32 + lane];
    float s = hv.x + hv.y + hv.z + hv.w;
#pragma unroll
    for (int off = 16; off > 0; off >>= 1) s += __shfl_xor(s, off, 32);
    float mu = s * (1.0f / HCH);
    f4 d = hv - mu;
    float vv = d.x * d.x + d.y * d.y + d.z * d.z + d.w * d.w;
#pragma unroll
    for (int off = 16; off > 0; off >>= 1) vv += __shfl_xor(vv, off, 32);
    float rs = rsqrtf(vv * (1.0f / HCH) + LN_EPS);
    const f4* g4 = (const f4*)g;
    const f4* b4 = (const f4*)b;
    f4 zz = d * rs * g4[lane] + b4[lane];
    zz.x = fmaxf(zz.x, 0.0f); zz.y = fmaxf(zz.y, 0.0f);
    zz.z = fmaxf(zz.z, 0.0f); zz.w = fmaxf(zz.w, 0.0f);
    ((f4*)z)[node * 32 + lane] = zz;
}

// ---------------- softmax aggregation (wave per node, CSR gather, no atomics) ----------------
__global__ __launch_bounds__(256) void agg_kernel(const float* __restrict__ z,
                                                  const int* __restrict__ rowptr,
                                                  const int* __restrict__ csrc,
                                                  const float* __restrict__ tptr, int layer,
                                                  float* __restrict__ o, int n) {
    int gid = blockIdx.x * blockDim.x + threadIdx.x;
    int node = gid >> 5;
    if (node >= n) return;
    int lane = gid & 31;
    float t = tptr[layer];
    int beg = rowptr[node], end = rowptr[node + 1];
    const f4* z4 = (const f4*)z;
    // pass 1: running max (all logits > 0 for t > 0; 0-init matches empty segments -> agg 0)
    f4 m = {0.0f, 0.0f, 0.0f, 0.0f};
    for (int e = beg; e < end; e++) {
        int s = csrc[e];
        f4 msg = z4[s * 32 + lane] + EPS_MSG;   // z already >= 0 (post-ReLU)
        f4 lg = msg * t;
        m.x = fmaxf(m.x, lg.x); m.y = fmaxf(m.y, lg.y);
        m.z = fmaxf(m.z, lg.z); m.w = fmaxf(m.w, lg.w);
    }
    // pass 2: exp-sum and weighted sum
    f4 den = {0, 0, 0, 0}, num = {0, 0, 0, 0};
    for (int e = beg; e < end; e++) {
        int s = csrc[e];
        f4 msg = z4[s * 32 + lane] + EPS_MSG;
        f4 lg = msg * t - m;
        f4 ex;
        ex.x = __expf(lg.x); ex.y = __expf(lg.y);
        ex.z = __expf(lg.z); ex.w = __expf(lg.w);
        den += ex;
        num += ex * msg;
    }
    f4 zn = z4[node * 32 + lane];
    f4 agg = num / (den + EPS_SM);
    ((f4*)o)[node * 32 + lane] = agg + zn;   // out = agg + z
}

// ---------------- transpose + f32->f16 weight conversion ----------------
// Wt[l][c][k] = W[l][k][c]
__global__ void convert_kernel(const float* __restrict__ W, half_t* __restrict__ Wt,
                               int K, int NCols, int total) {
    int i = blockIdx.x * blockDim.x + threadIdx.x;
    if (i >= total) return;
    int c = i % NCols;
    int k = (i / NCols) % K;
    int l = i / (NCols * K);
    Wt[((size_t)l * NCols + c) * K + k] = (half_t)W[i];
}

// ---------------- fused update MLP ----------------
// TDM-stage 32x128 f32 tile -> f16 -> WMMA GEMM1 -> bias -> LN -> ReLU -> WMMA GEMM2 -> residual
__global__ __launch_bounds__(256) void mlp_kernel(const float* __restrict__ o, float* __restrict__ h,
                                                  const half_t* __restrict__ w1t, const float* __restrict__ b1,
                                                  const float* __restrict__ g1, const float* __restrict__ bb1,
                                                  const half_t* __restrict__ w2t, const float* __restrict__ b2,
                                                  int n) {
    const int AJ = 136;   // 16B-aligned padded LDS row stride (halves)
    const int HJ = 264;
    __shared__ __align__(16) float  sC[MROWS * HIDC];  // staging (f32 tile) then GEMM1 output
    __shared__ __align__(16) half_t sA[MROWS * 136];   // activations f16 [32 x 128]
    __shared__ __align__(16) half_t sH[MROWS * 264];   // hidden f16 [32 x 256]
    int tid = threadIdx.x;
    int lane = tid & 31, wave = tid >> 5;
    int n0 = blockIdx.x * MROWS;
    int rowsRem = n - n0;                  // > 0 by grid construction

    // ---- stage A tile (rowsRem-clamped) into sC via Tensor Data Mover ----
    if (wave == 0) {
        unsigned lds_off = (unsigned)(uintptr_t)(&sC[0]);     // low 32b of flat = LDS byte offset
        unsigned long long ga = (unsigned long long)(uintptr_t)(o + (size_t)n0 * HCH);
        unsigned tdim1 = (unsigned)(rowsRem < MROWS ? rowsRem : MROWS);
        // D# group0: count=1 | lds_addr | global_addr[56:0] | type=2
        v4u g0;
        g0.x = 1u;
        g0.y = lds_off;
        g0.z = (unsigned)(ga & 0xFFFFFFFFu);
        g0.w = (unsigned)((ga >> 32) & 0x01FFFFFFu) | 0x80000000u;
        // D# group1: data_size=2 (4B); tensor_dim0=128; tensor_dim1=tdim1 (OOB rows -> 0);
        //            tile_dim0=128; tile_dim1=32; tensor_dim0_stride=128
        v8i g1v;
        g1v[0] = 0x00020000;                                   // data_size=4B
        g1v[1] = (int)((HCH & 0xFFFF) << 16);                  // tensor_dim0 lo16 @ bits48-63
        g1v[2] = (int)(((HCH >> 16) & 0xFFFF) | ((tdim1 & 0xFFFF) << 16));
        g1v[3] = (int)(((tdim1 >> 16) & 0xFFFF) | (HCH << 16));// tile_dim0=128 @ bits112-127
        g1v[4] = MROWS;                                        // tile_dim1=32, tile_dim2=0
        g1v[5] = HCH;                                          // tensor_dim0_stride lo32
        g1v[6] = 0;
        g1v[7] = 0;
        v4i gz = {0, 0, 0, 0};                                 // 2D tensor: groups 2/3 unused
        v8i gz8 = {0, 0, 0, 0, 0, 0, 0, 0};                    // extra group (6-arg form): zero
        __builtin_amdgcn_tensor_load_to_lds(g0, g1v, gz, gz, gz8, 0);
        __builtin_amdgcn_s_wait_tensorcnt(0);
    }
    __syncthreads();

    // ---- convert staged f32 tile -> padded f16 A buffer ----
    {
        int r = tid >> 3;               // 0..31
        int c0 = (tid & 7) * 16;        // 0..112
        bool live = (n0 + r) < n;
#pragma unroll
        for (int j = 0; j < 16; j++) {
            float v = live ? sC[r * HCH + c0 + j] : 0.0f;
            sA[r * AJ + c0 + j] = (half_t)v;
        }
    }
    __syncthreads();

    int row = lane & 15, hi = lane >> 4;
    int kA = hi * 8;    // A-fragment K base (ISA 16-bit A 16x32 layout)
    int kB = hi * 8;    // B-fragment K base

    // A fragments for both M-tiles, all K-steps
    v16h a[2][4];
#pragma unroll
    for (int mt = 0; mt < 2; mt++) {
#pragma unroll
        for (int ks = 0; ks < 4; ks++) {
            const half_t* p = &sA[(mt * 16 + row) * AJ + ks * 32 + kA];
            v8h lo = *(const v8h*)p;
            v8h hi8 = *(const v8h*)(p + 16);
            a[mt][ks] = __builtin_shufflevector(lo, hi8, 0, 1, 2, 3, 4, 5, 6, 7, 8, 9, 10, 11, 12, 13, 14, 15);
        }
    }
    // GEMM1: [32x128] x [128x256]; 8 waves x 2 N-tiles x 2 M-tiles x 4 ksteps = 16 wmma/wave
#pragma unroll
    for (int tnum = 0; tnum < 2; tnum++) {
        int ncol = (wave + tnum * 8) * 16 + row;
        v8f acc0 = {}, acc1 = {};
#pragma unroll
        for (int ks = 0; ks < 4; ks++) {
            const half_t* bp = &w1t[(size_t)ncol * HCH + ks * 32 + kB];
            v8h blo = *(const v8h*)bp;
            v8h bhi = *(const v8h*)(bp + 16);
            v16h bf = __builtin_shufflevector(blo, bhi, 0, 1, 2, 3, 4, 5, 6, 7, 8, 9, 10, 11, 12, 13, 14, 15);
            acc0 = __builtin_amdgcn_wmma_f32_16x16x32_f16(false, a[0][ks], false, bf,
                                                          (short)0, acc0, false, false);
            acc1 = __builtin_amdgcn_wmma_f32_16x16x32_f16(false, a[1][ks], false, bf,
                                                          (short)0, acc1, false, false);
        }
        float bias = b1[ncol];
#pragma unroll
        for (int r8 = 0; r8 < 8; r8++) {
            sC[(hi * 8 + r8) * HIDC + ncol] = acc0[r8] + bias;
            sC[(16 + hi * 8 + r8) * HIDC + ncol] = acc1[r8] + bias;
        }
    }
    __syncthreads();
    // row-wise LayerNorm(256) + ReLU -> f16; wave handles rows wave+{0,8,16,24}
#pragma unroll
    for (int rr = 0; rr < 4; rr++) {
        int r = wave + rr * 8;
        const float* cr = &sC[r * HIDC];
        float s = 0.0f;
#pragma unroll
        for (int j = 0; j < 8; j++) s += cr[lane * 8 + j];
#pragma unroll
        for (int off = 16; off > 0; off >>= 1) s += __shfl_xor(s, off, 32);
        float mu = s * (1.0f / HIDC);
        float vv = 0.0f;
#pragma unroll
        for (int j = 0; j < 8; j++) { float d = cr[lane * 8 + j] - mu; vv += d * d; }
#pragma unroll
        for (int off = 16; off > 0; off >>= 1) vv += __shfl_xor(vv, off, 32);
        float rs = rsqrtf(vv * (1.0f / HIDC) + LN_EPS);
#pragma unroll
        for (int j = 0; j < 8; j++) {
            int c = lane * 8 + j;
            float val = (cr[c] - mu) * rs * g1[c] + bb1[c];
            sH[r * HJ + c] = (half_t)fmaxf(val, 0.0f);
        }
    }
    __syncthreads();
    // GEMM2: [32x256] x [256x128]; 8 waves x 1 N-tile x 2 M-tiles x 8 ksteps = 16 wmma/wave
    int ncol2 = wave * 16 + row;
    v8f acc20 = {}, acc21 = {};
#pragma unroll
    for (int ks = 0; ks < 8; ks++) {
        const half_t* bp = &w2t[(size_t)ncol2 * HIDC + ks * 32 + kB];
        v8h blo = *(const v8h*)bp;
        v8h bhi = *(const v8h*)(bp + 16);
        v16h bf = __builtin_shufflevector(blo, bhi, 0, 1, 2, 3, 4, 5, 6, 7, 8, 9, 10, 11, 12, 13, 14, 15);
        const half_t* p0 = &sH[row * HJ + ks * 32 + kA];
        v8h lo0 = *(const v8h*)p0;
        v8h hi0 = *(const v8h*)(p0 + 16);
        v16h af0 = __builtin_shufflevector(lo0, hi0, 0, 1, 2, 3, 4, 5, 6, 7, 8, 9, 10, 11, 12, 13, 14, 15);
        acc20 = __builtin_amdgcn_wmma_f32_16x16x32_f16(false, af0, false, bf,
                                                       (short)0, acc20, false, false);
        const half_t* p1 = &sH[(16 + row) * HJ + ks * 32 + kA];
        v8h lo1 = *(const v8h*)p1;
        v8h hi1 = *(const v8h*)(p1 + 16);
        v16h af1 = __builtin_shufflevector(lo1, hi1, 0, 1, 2, 3, 4, 5, 6, 7, 8, 9, 10, 11, 12, 13, 14, 15);
        acc21 = __builtin_amdgcn_wmma_f32_16x16x32_f16(false, af1, false, bf,
                                                       (short)0, acc21, false, false);
    }
    float bias2 = b2[ncol2];
#pragma unroll
    for (int r8 = 0; r8 < 8; r8++) {
        int nn0 = n0 + hi * 8 + r8;
        if (nn0 < n) h[(size_t)nn0 * HCH + ncol2] += acc20[r8] + bias2;
        int nn1 = n0 + 16 + hi * 8 + r8;
        if (nn1 < n) h[(size_t)nn1 * HCH + ncol2] += acc21[r8] + bias2;
    }
}

// ---------------- global add pool + classifier ----------------
__global__ __launch_bounds__(256) void pool_kernel(const float* __restrict__ h, const int* __restrict__ batch,
                                                   float* __restrict__ pooled, int n) {
    int gid = blockIdx.x * blockDim.x + threadIdx.x;
    int node = gid >> 5;
    if (node >= n) return;
    int lane = gid & 31;
    int b = batch[node];
    f4 hv = ((const f4*)h)[node * 32 + lane];
    float* pb = &pooled[b * HCH + lane * 4];
    atomicAdd(pb + 0, hv.x);
    atomicAdd(pb + 1, hv.y);
    atomicAdd(pb + 2, hv.z);
    atomicAdd(pb + 3, hv.w);
}

__global__ void cls_kernel(const float* __restrict__ pooled, const float* __restrict__ W,
                           const float* __restrict__ b, float* __restrict__ out) {
    int i = blockIdx.x * blockDim.x + threadIdx.x;
    if (i >= NGR * NCL) return;
    int g = i >> 1, c = i & 1;
    float acc = b[c];
#pragma unroll 8
    for (int k = 0; k < HCH; k++) acc += pooled[g * HCH + k] * W[k * NCL + c];
    out[i] = acc;
}

// ---------------- host launcher ----------------
extern "C" void kernel_launch(void* const* d_in, const int* in_sizes, int n_in,
                              void* d_out, int out_size, void* d_ws, size_t ws_size,
                              hipStream_t stream) {
    const float* x    = (const float*)d_in[0];
    const int*   ei   = (const int*)  d_in[1];
    const int*   batch= (const int*)  d_in[2];
    const float* encW = (const float*)d_in[3];
    const float* encb = (const float*)d_in[4];
    const float* t    = (const float*)d_in[5];
    const float* W1   = (const float*)d_in[6];
    const float* b1   = (const float*)d_in[7];
    const float* g1   = (const float*)d_in[8];
    const float* bb1  = (const float*)d_in[9];
    const float* W2   = (const float*)d_in[10];
    const float* b2   = (const float*)d_in[11];
    const float* lng  = (const float*)d_in[12];
    const float* lnb  = (const float*)d_in[13];
    const float* clsW = (const float*)d_in[14];
    const float* clsb = (const float*)d_in[15];
    float* out = (float*)d_out;

    const int n = in_sizes[2];          // N nodes
    const int e = in_sizes[1] / 2;      // E edges
    const int L = in_sizes[5];          // layers
    const int* src = ei;
    const int* dst = ei + e;

    // workspace carve-out
    char* ws = (char*)d_ws;
    size_t off = 0;
    auto carve = [&](size_t bytes) -> char* {
        char* p = ws + off;
        off = (off + bytes + 255) & ~(size_t)255;
        return p;
    };
    float*  h      = (float*) carve((size_t)n * HCH * 4);
    float*  z      = (float*) carve((size_t)n * HCH * 4);
    float*  o      = (float*) carve((size_t)n * HCH * 4);
    float*  pooled = (float*) carve((size_t)NGR * HCH * 4);
    half_t* w1t    = (half_t*)carve((size_t)L * HCH * HIDC * 2);
    half_t* w2t    = (half_t*)carve((size_t)L * HCH * HIDC * 2);
    int*    rowptr = (int*)   carve((size_t)(n + 1) * 4);
    int*    csrc   = (int*)   carve((size_t)e * 4);
    int*    cnt    = (int*)   carve((size_t)n * 4);
    int*    cur    = (int*)   carve((size_t)n * 4);

    const int TB = 256;
    // CSR build (deterministic, rebuilt every call)
    zero_i32<<<(n + TB - 1) / TB, TB, 0, stream>>>(cnt, n);
    zero_i32<<<(n + TB - 1) / TB, TB, 0, stream>>>(cur, n);
    count_kernel<<<(e + TB - 1) / TB, TB, 0, stream>>>(dst, cnt, e);
    scan_kernel<<<1, 1024, 0, stream>>>(cnt, rowptr, n);
    fill_kernel<<<(e + TB - 1) / TB, TB, 0, stream>>>(src, dst, rowptr, cur, csrc, e);

    // encoder + weight conversion (once per call)
    encoder_kernel<<<((size_t)n * HCH + TB - 1) / TB, TB, 0, stream>>>(x, encW, encb, h, n);
    {
        int tot1 = L * HCH * HIDC;
        convert_kernel<<<(tot1 + TB - 1) / TB, TB, 0, stream>>>(W1, w1t, HCH, HIDC, tot1);
        int tot2 = L * HIDC * HCH;
        convert_kernel<<<(tot2 + TB - 1) / TB, TB, 0, stream>>>(W2, w2t, HIDC, HCH, tot2);
    }

    // DeepGCN layers
    int waveGrid = (n + 7) / 8;           // 8 waves (nodes) per 256-thread block
    int mlpGrid = (n + MROWS - 1) / MROWS;
    for (int l = 0; l < L; l++) {
        znorm_kernel<<<waveGrid, TB, 0, stream>>>(h, z, lng + l * HCH, lnb + l * HCH, n);
        agg_kernel<<<waveGrid, TB, 0, stream>>>(z, rowptr, csrc, t, l, o, n);
        mlp_kernel<<<mlpGrid, TB, 0, stream>>>(o, h,
                                               w1t + (size_t)l * HCH * HIDC, b1 + l * HIDC,
                                               g1 + l * HIDC, bb1 + l * HIDC,
                                               w2t + (size_t)l * HCH * HIDC, b2 + l * HCH, n);
    }

    // pool + classify
    zero_f32<<<(NGR * HCH + TB - 1) / TB, TB, 0, stream>>>(pooled, NGR * HCH);
    pool_kernel<<<waveGrid, TB, 0, stream>>>(h, batch, pooled, n);
    cls_kernel<<<(NGR * NCL + TB - 1) / TB, TB, 0, stream>>>(pooled, clsW, clsb, out);
}
